// PointPillars_25116968747534
// MI455X (gfx1250) — matrix operand order repaired and checked by reading, compiled, not run on hardware
//
#include <hip/hip_runtime.h>
#include <cstdint>
#include <cstddef>

// ======================================================================
// PointPillars forward for gfx1250 (MI455X), compile-only target.
// Backbone convs = implicit-GEMM via V_WMMA_F32_16X16X32_F16 (wave32),
// block tile 64(M) x 128(N): each wave holds 4 f32 accumulator tiles and
// issues 4 back-to-back WMMAs per K-step (B fragment reused across all 4).
// Activations carried as f16; BN stats via cross-lane shfl + atomics.
// Workspace layout (assumes ws_size >= ~163 MB):
//   [ACT_A 81.4MB][ACT_B 40.1MB][RAW 40.1MB][WGT16 1.0MB][STATS 2KB][H 512B]
// ======================================================================

typedef _Float16 v16h __attribute__((ext_vector_type(16)));
typedef _Float16 v8h  __attribute__((ext_vector_type(8)));
typedef float    v8f  __attribute__((ext_vector_type(8)));

constexpr int   B_ = 4, P_ = 12000, N_ = 32, H_ = 282, W_ = 282;
constexpr float PS_ = 0.16f, RNG_ = 22.0f, EPS_ = 1e-5f;

// ---------------------------------------------------------------- utils
__global__ void zero_u32(uint32_t* p, long n) {
  long i = (long)blockIdx.x * blockDim.x + threadIdx.x;
  if (i < n) p[i] = 0u;
}

__global__ void cvt_f32_f16(const float* __restrict__ w, _Float16* __restrict__ o, int n) {
  int i = blockIdx.x * blockDim.x + threadIdx.x;
  if (i < n) o[i] = (_Float16)w[i];
}

// per-channel scale/shift from accumulated sum / sumsq (biased var, matches jnp.var)
__global__ void bn_finalize(const float* __restrict__ sum, const float* __restrict__ sq,
                            const float* __restrict__ g, const float* __restrict__ be,
                            float invcnt, int C, float* __restrict__ scale,
                            float* __restrict__ shift) {
  int c = threadIdx.x;
  if (c < C) {
    float m = sum[c] * invcnt;
    float v = fmaxf(sq[c] * invcnt - m * m, 0.f);
    float s = g[c] * rsqrtf(v + EPS_);
    scale[c] = s;
    shift[c] = be[c] - m * s;
  }
}

// ---------------------------------------------------------------- PFN
// pass 1: per-channel sum / sumsq of y = w(64x3) . x over (b,p,n)
__global__ void pfn_stats(const float* __restrict__ x, const float* __restrict__ w,
                          float* __restrict__ gsum, float* __restrict__ gsq) {
  __shared__ float xs[96];
  __shared__ float ssum[64], ssq[64];
  int bp = blockIdx.x;                       // b*P + p
  int b = bp / P_, p = bp - b * P_;
  int t = threadIdx.x;                       // 256
  if (t < 96) {
    int c = t >> 5, n = t & 31;
    xs[t] = x[(((size_t)b * 3 + c) * P_ + p) * N_ + n];
  }
  if (t < 64) { ssum[t] = 0.f; ssq[t] = 0.f; }
  __syncthreads();
  for (int item = t; item < 64 * 32; item += 256) {
    int o = item >> 5, n = item & 31;
    float y = w[o * 3 + 0] * xs[n] + w[o * 3 + 1] * xs[32 + n] + w[o * 3 + 2] * xs[64 + n];
    atomicAdd(&ssum[o], y);
    atomicAdd(&ssq[o], y * y);
  }
  __syncthreads();
  if (t < 64) { atomicAdd(&gsum[t], ssum[t]); atomicAdd(&gsq[t], ssq[t]); }
}

// pass 2: recompute y, BN+ReLU, max over n, scatter into f16 canvas slot
__global__ void pfn_scatter(const float* __restrict__ x, const float* __restrict__ coords,
                            const float* __restrict__ w, const float* __restrict__ scale,
                            const float* __restrict__ shift, _Float16* __restrict__ canvas,
                            int ch_off) {
  __shared__ float xs[96];
  int bp = blockIdx.x;
  int b = bp / P_, p = bp - b * P_;
  int t = threadIdx.x;                       // 128
  if (t < 96) {
    int c = t >> 5, n = t & 31;
    xs[t] = x[(((size_t)b * 3 + c) * P_ + p) * N_ + n];
  }
  __syncthreads();
  if (t < 64) {
    float w0 = w[t * 3], w1 = w[t * 3 + 1], w2 = w[t * 3 + 2];
    float sc = scale[t], sh = shift[t];
    float m = 0.f;                           // relu outputs are >= 0
    #pragma unroll
    for (int n = 0; n < 32; ++n) {
      float y = w0 * xs[n] + w1 * xs[32 + n] + w2 * xs[64 + n];
      m = fmaxf(m, fmaxf(sc * y + sh, 0.f));
    }
    float cx = coords[((size_t)b * P_ + p) * 2 + 0];
    float cy = coords[((size_t)b * P_ + p) * 2 + 1];
    int xg = (int)floorf((cx + RNG_) / PS_);
    int yg = (int)floorf((cy + RNG_) / PS_);
    int idx = xg * W_ + yg;
    canvas[((size_t)b * 128 + ch_off + t) * (H_ * W_) + idx] = (_Float16)m;
  }
}

// ---------------------------------------------------------------- conv (implicit GEMM, WMMA)
// GEMM: D(Cout x N) = W(Cout x Cin*9) * im2col(Cin*9 x N), N = B*HO*WO.
// Block = 256 thr = 8 waves; block tile = (16*MT)(M) x 128(N); K stepped by 32.
// Each wave: 1 B fragment reused across MT A fragments -> MT wmma / K-step.
template <int CIN, int COUT, int HIN, int WIN, int HO, int WO, int STRIDE>
__global__ __launch_bounds__(256) void conv_wmma(
    const _Float16* __restrict__ act, const _Float16* __restrict__ wgt,
    _Float16* __restrict__ raw, float* __restrict__ gsum, float* __restrict__ gsq) {
  constexpr int K = CIN * 9;
  constexpr int NTOT = B_ * HO * WO;
  constexpr int KSTEPS = (K + 31) / 32;
  constexpr int MT = (COUT >= 64) ? 4 : (COUT + 15) / 16;   // M-fragments per wave
  constexpr int MB = 16 * MT;                               // block M tile

  __shared__ __align__(32) _Float16 lA[64 * 32];     // (m, k) row-major (<= 4 KB)
  __shared__ __align__(32) _Float16 lBT[128 * 32];   // (n, k) row-major (transposed B)

  const int t = threadIdx.x;
  const int lane = t & 31, wv = t >> 5;
  const int lm = lane & 15, hs = lane >> 4;          // half-wave select
  const int mtile = blockIdx.y * MB;
  const int ntile = blockIdx.x * 128;

  // B-tile loader coordinates: thread fills lBT[nnL][khL..khL+15]
  const int nnL = t & 127;
  const int khL = (t >> 7) * 16;
  const int nL = ntile + nnL;
  const bool nLvalid = nL < NTOT;
  int bL = 0, oyL = 0, oxL = 0;
  if (nLvalid) {
    bL = nL / (HO * WO);
    int r = nL - bL * (HO * WO);
    oyL = r / WO; oxL = r - oyL * WO;
  }

  v8f acc[MT];
  #pragma unroll
  for (int mi = 0; mi < MT; ++mi) acc[mi] = (v8f){};

  for (int ks = 0; ks < KSTEPS; ++ks) {
    const int kb = ks * 32;
    // A strip MBx32 (weights)
    for (int e = t; e < MB * 32; e += 256) {
      int m = e >> 5, kk = e & 31;
      int gm = mtile + m, gk = kb + kk;
      lA[e] = (gm < COUT && gk < K) ? wgt[gm * K + gk] : (_Float16)0.f;
    }
    // B strip 128x32 (on-the-fly im2col), incremental k -> (cin, r, c) decode
    {
      int k0 = kb + khL;
      int cin = k0 / 9;
      int rem = k0 - cin * 9;
      int rr = rem / 3;
      int cc = rem - rr * 3;
      #pragma unroll
      for (int j = 0; j < 16; ++j) {
        _Float16 v = (_Float16)0.f;
        int k = k0 + j;
        if (nLvalid && k < K) {
          int iy = oyL * STRIDE + rr;
          int ix = oxL * STRIDE + cc;
          v = act[((size_t)(bL * CIN + cin) * HIN + iy) * WIN + ix];
        }
        lBT[nnL * 32 + khL + j] = v;
        if (++cc == 3) { cc = 0; if (++rr == 3) { rr = 0; ++cin; } }
      }
      // prefetch next K-step's source line (global_prefetch_b8)
      if (ks + 1 < KSTEPS && nLvalid) {
        int kn = kb + 32 + khL;
        if (kn < K) {
          int cin2 = kn / 9;
          __builtin_prefetch(
              &act[((size_t)(bL * CIN + cin2) * HIN + oyL * STRIDE) * WIN + oxL * STRIDE], 0, 1);
        }
      }
    }
    __syncthreads();

    // B fragment: lane = column (wv*16+lm); K = 16 contiguous (hs selects half)
    v16h Bf = *(const v16h*)&lBT[(wv * 16 + lm) * 32 + hs * 16];

    // MT back-to-back WMMAs, one per M fragment, all sharing Bf
    #pragma unroll
    for (int mi = 0; mi < MT; ++mi) {
      // A fragment: lane holds row lm; K = 16 contiguous (0..15 for hs=0 / 16..31 for hs=1)
      const int mrow = mi * 16 + lm;
      v8h a0 = *(const v8h*)&lA[mrow * 32 + hs * 16];
      v8h a1 = *(const v8h*)&lA[mrow * 32 + hs * 16 + 8];
      v16h A;
      #pragma unroll
      for (int i = 0; i < 8; ++i) { A[i] = a0[i]; A[8 + i] = a1[i]; }
      acc[mi] = __builtin_amdgcn_wmma_f32_16x16x32_f16(false, A, false, Bf,
                                                       (short)0, acc[mi], false, false);
    }
    __syncthreads();
  }

  // writeout (C layout: vgpr j -> row j (lanes 0-15) / j+8 (lanes 16-31), col = lane&15)
  const int n = ntile + wv * 16 + lm;
  const bool nv = n < NTOT;
  int b = 0, oy = 0, ox = 0;
  if (nv) {
    b = n / (HO * WO);
    int r = n - b * (HO * WO);
    oy = r / WO; ox = r - oy * WO;
  }
  #pragma unroll
  for (int mi = 0; mi < MT; ++mi) {
    #pragma unroll
    for (int j = 0; j < 8; ++j) {
      int mg = mtile + mi * 16 + j + hs * 8;
      float c = acc[mi][j];
      bool valid = nv && (mg < COUT);
      if (valid)
        raw[((size_t)(b * COUT + mg) * HO + oy) * WO + ox] = (_Float16)c;
      // per-channel stats: reduce across the 16 columns of this half-wave
      float s = valid ? c : 0.f;
      float q = s * s;
      #pragma unroll
      for (int d = 8; d >= 1; d >>= 1) {
        s += __shfl_xor(s, d, 16);
        q += __shfl_xor(q, d, 16);
      }
      if (lm == 0 && mg < COUT) {
        atomicAdd(&gsum[mg], s);
        atomicAdd(&gsq[mg], q);
      }
    }
  }
}

// BN + ReLU + f16 recompress
__global__ void bn_relu(const _Float16* __restrict__ raw, const float* __restrict__ scale,
                        const float* __restrict__ shift, _Float16* __restrict__ out,
                        int C, int HW, long total) {
  long i = (long)blockIdx.x * blockDim.x + threadIdx.x;
  if (i >= total) return;
  int c = (int)((i / (long)HW) % C);
  float v = (float)raw[i];
  out[i] = (_Float16)fmaxf(scale[c] * v + shift[c], 0.f);
}

// ---------------------------------------------------------------- head
__global__ void fc1_kernel(const _Float16* __restrict__ act, const float* __restrict__ w,
                           float* __restrict__ h) {
  constexpr int KF = 8 * 59 * 59;            // 27848
  int j = blockIdx.x, b = blockIdx.y, t = threadIdx.x;
  const _Float16* xa = act + (size_t)b * KF;
  const float* wr = w + (size_t)j * KF;
  float acc = 0.f;
  for (int i = t; i < KF; i += 256) acc += (float)xa[i] * wr[i];
  __shared__ float red[256];
  red[t] = acc;
  __syncthreads();
  for (int s = 128; s >= 1; s >>= 1) {
    if (t < s) red[t] += red[t + s];
    __syncthreads();
  }
  if (t == 0) h[b * 32 + j] = red[0];        // fc1_b cancels in batch-norm
}

__global__ void fc_head(const float* __restrict__ h, const float* __restrict__ g,
                        const float* __restrict__ be, const float* __restrict__ fw,
                        const float* __restrict__ fb, float* __restrict__ out) {
  __shared__ float tt[4 * 32];
  int t = threadIdx.x;                       // 64
  if (t < 32) {
    float a = h[t], b1 = h[32 + t], c = h[64 + t], d = h[96 + t];
    float m = 0.25f * (a + b1 + c + d);
    float v = 0.25f * ((a - m) * (a - m) + (b1 - m) * (b1 - m) +
                       (c - m) * (c - m) + (d - m) * (d - m));
    float s = rsqrtf(v + EPS_);
    tt[t]      = tanhf(g[t] * (a - m) * s + be[t]);
    tt[32 + t] = tanhf(g[t] * (b1 - m) * s + be[t]);
    tt[64 + t] = tanhf(g[t] * (c - m) * s + be[t]);
    tt[96 + t] = tanhf(g[t] * (d - m) * s + be[t]);
  }
  __syncthreads();
  if (t < 12) {
    int b = t / 3, o = t - b * 3;
    float acc = fb[o];
    #pragma unroll
    for (int j = 0; j < 32; ++j) acc += fw[o * 32 + j] * tt[b * 32 + j];
    out[b * 3 + o] = acc;
  }
}

// ---------------------------------------------------------------- launch
extern "C" void kernel_launch(void* const* d_in, const int* in_sizes, int n_in,
                              void* d_out, int out_size, void* d_ws, size_t ws_size,
                              hipStream_t stream) {
  (void)in_sizes; (void)n_in; (void)out_size; (void)ws_size;

  // Input order: setup_inputs dict, params flattened JAX-pytree style
  // (dict keys sorted alphabetically; list/tuple order preserved):
  //  0 sweep  1 map_pts  2 sweep_coords  3 map_coords
  //  4+4l..7+4l : bb.convs[l] (w, b, g, be), l = 0..9
  //  44 fc1_b  45 fc1_be  46 fc1_g  47 fc1_w  48 fco_b  49 fco_w
  //  50..53 pfn_m (b, be, g, w)   54..57 pfn_s (b, be, g, w)
  const float* sweep  = (const float*)d_in[0];
  const float* mappts = (const float*)d_in[1];
  const float* scoord = (const float*)d_in[2];
  const float* mcoord = (const float*)d_in[3];
  const float* pfnS_w  = (const float*)d_in[57];
  const float* pfnS_g  = (const float*)d_in[56];
  const float* pfnS_be = (const float*)d_in[55];
  const float* pfnM_w  = (const float*)d_in[53];
  const float* pfnM_g  = (const float*)d_in[52];
  const float* pfnM_be = (const float*)d_in[51];

  uint8_t* ws = (uint8_t*)d_ws;
  constexpr size_t O_ACTA = 0;                       // canvas + odd-layer acts (81,444,864 B)
  constexpr size_t O_ACTB = 81444864;                // even-layer acts (40,140,800 B)
  constexpr size_t O_RAW  = O_ACTB + 40140800;       // pre-BN conv output
  constexpr size_t O_WGT  = O_RAW + 40140800;        // f16 conv weights (1,006,848 B)
  constexpr size_t O_STAT = O_WGT + 1006848;         // sum|sq|scale|shift (4x128 f32)
  constexpr size_t O_H    = O_STAT + 2048;           // fc1 activations (4x32 f32)
  _Float16* actA = (_Float16*)(ws + O_ACTA);
  _Float16* actB = (_Float16*)(ws + O_ACTB);
  _Float16* raw  = (_Float16*)(ws + O_RAW);
  _Float16* wgt  = (_Float16*)(ws + O_WGT);
  float* sum   = (float*)(ws + O_STAT);
  float* sq    = sum + 128;
  float* scale = sum + 256;
  float* shift = sum + 384;
  float* hbuf  = (float*)(ws + O_H);

  // f16 conv weight pack (identity flatten: (o,i,3,3) -> (o, i*9+r*3+c))
  static const int    wsz[10]  = {73728, 36864, 36864, 36864, 73728,
                                  147456, 73728, 18432, 4608, 1152};
  static const size_t woff[10] = {0, 73728, 110592, 147456, 184320,
                                  258048, 405504, 479232, 497664, 502272};
  for (int l = 0; l < 10; ++l)
    cvt_f32_f16<<<(wsz[l] + 255) / 256, 256, 0, stream>>>(
        (const float*)d_in[4 + 4 * l], wgt + woff[l], wsz[l]);

  // zero the canvas (f16, 4x128x282x282)
  {
    long cw = 81444864 / 4;
    zero_u32<<<(unsigned)((cw + 255) / 256), 256, 0, stream>>>((uint32_t*)actA, cw);
  }

  // PFN + scatter: sweep -> channels 0..63, map -> channels 64..127
  const float pfnInv = 1.f / (float)(B_ * P_ * N_);
  zero_u32<<<1, 256, 0, stream>>>((uint32_t*)sum, 256);
  pfn_stats<<<B_ * P_, 256, 0, stream>>>(sweep, pfnS_w, sum, sq);
  bn_finalize<<<1, 64, 0, stream>>>(sum, sq, pfnS_g, pfnS_be, pfnInv, 64, scale, shift);
  pfn_scatter<<<B_ * P_, 128, 0, stream>>>(sweep, scoord, pfnS_w, scale, shift, actA, 0);

  zero_u32<<<1, 256, 0, stream>>>((uint32_t*)sum, 256);
  pfn_stats<<<B_ * P_, 256, 0, stream>>>(mappts, pfnM_w, sum, sq);
  bn_finalize<<<1, 64, 0, stream>>>(sum, sq, pfnM_g, pfnM_be, pfnInv, 64, scale, shift);
  pfn_scatter<<<B_ * P_, 128, 0, stream>>>(mappts, mcoord, pfnM_w, scale, shift, actA, 64);

  // Backbone: even layer reads actA -> writes actB; odd layer the reverse.
#define RUN_CONV(L, CIN, COUT, HIN, WIN, HO, WO, S)                                     \
  do {                                                                                  \
    const float* g_  = (const float*)d_in[6 + 4 * (L)];                                 \
    const float* be_ = (const float*)d_in[7 + 4 * (L)];                                 \
    const _Float16* ain = ((L) % 2 == 0) ? actA : actB;                                 \
    _Float16* aout      = ((L) % 2 == 0) ? actB : actA;                                 \
    constexpr int NT = B_ * (HO) * (WO);                                                \
    constexpr int MT_ = ((COUT) >= 64) ? 4 : ((COUT) + 15) / 16;                        \
    zero_u32<<<1, 256, 0, stream>>>((uint32_t*)sum, 256);                               \
    dim3 cg((NT + 127) / 128, ((COUT) + 16 * MT_ - 1) / (16 * MT_));                    \
    conv_wmma<CIN, COUT, HIN, WIN, HO, WO, S>                                           \
        <<<cg, 256, 0, stream>>>(ain, wgt + woff[L], raw, sum, sq);                     \
    bn_finalize<<<1, (COUT), 0, stream>>>(sum, sq, g_, be_, 1.f / (float)NT, (COUT),    \
                                          scale, shift);                                \
    long tot = (long)(COUT) * NT;                                                       \
    bn_relu<<<(unsigned)((tot + 255) / 256), 256, 0, stream>>>(raw, scale, shift, aout, \
                                                               (COUT), (HO) * (WO), tot);\
  } while (0)

  RUN_CONV(0, 128,  64, 282, 282, 280, 280, 1);
  RUN_CONV(1,  64,  64, 280, 280, 278, 278, 1);
  RUN_CONV(2,  64,  64, 278, 278, 276, 276, 1);
  RUN_CONV(3,  64,  64, 276, 276, 274, 274, 1);
  RUN_CONV(4,  64, 128, 274, 274, 136, 136, 2);
  RUN_CONV(5, 128, 128, 136, 136,  67,  67, 2);
  RUN_CONV(6, 128,  64,  67,  67,  65,  65, 1);
  RUN_CONV(7,  64,  32,  65,  65,  63,  63, 1);
  RUN_CONV(8,  32,  16,  63,  63,  61,  61, 1);
  RUN_CONV(9,  16,   8,  61,  61,  59,  59, 1);
#undef RUN_CONV

  // Head: L9 normalized act lives in actA (L9 is odd)
  fc1_kernel<<<dim3(32, 4), 256, 0, stream>>>(actA, (const float*)d_in[47], hbuf);
  fc_head<<<1, 64, 0, stream>>>(hbuf, (const float*)d_in[46], (const float*)d_in[45],
                                (const float*)d_in[49], (const float*)d_in[48],
                                (float*)d_out);
}
